// DeepStatisticalSolver2_13297218749113
// MI455X (gfx1250) — matrix operand order repaired and checked by reading, not compiled, not run on hardware
//
#include <hip/hip_runtime.h>
#include <hip/hip_bf16.h>

// ---------------------------------------------------------------------------
// Deep statistical solver GNN forward on gfx1250 (MI455X), all GEMMs on
// v_wmma_f32_16x16x32_bf16. Wave32 / WGP model throughout.
// ---------------------------------------------------------------------------

typedef __bf16 bf16;
typedef __attribute__((ext_vector_type(16))) __bf16 v16bf;
typedef __attribute__((ext_vector_type(8)))  float  v8f;

// Problem constants (from the reference)
#define Bc   32
#define Nc   2000
#define Ec   2600
#define Lc   64
#define DAc  6
#define DBc  8
#define DUc  2
#define DTc  0.2f
#define KE   199   // 1 + 3*64 + 6
#define KEP  224   // padded to multiple of 32
#define KB   139   // 1 + 2*64 + 2 + 8
#define KBP  160
#define RE   (Bc * Ec)   // 83200 edge rows
#define RB   (Bc * Nc)   // 64000 node rows

// ---------------------------------------------------------------------------
// One MLP layer as WMMA. A: LDS tile [64][K] bf16 row-major.
// Wt: global [64][K] bf16 (row = output column n, contiguous K).
// Each wave computes two 16x16 output tiles of the 64x64 block result.
// ---------------------------------------------------------------------------
template <int K>
__device__ __forceinline__ void mlp_layer(const bf16* __restrict__ A,
                                          const bf16* __restrict__ Wt,
                                          const float* __restrict__ bias,
                                          bf16* __restrict__ Hout,   // LDS [64][64] or nullptr
                                          float* __restrict__ Yout,  // global when Hout==nullptr
                                          long row_base, long rows,
                                          bool do_tanh) {
  const int lane = threadIdx.x & 31;
  const int wave = threadIdx.x >> 5;
  const int mt   = wave >> 1;                 // M tile 0..3
  const int nb0  = (wave & 1) * 32;           // two N tiles per wave
  const int nb1  = nb0 + 16;
  const int mrow = mt * 16 + (lane & 15);     // A-fragment source row
  const int koffA = (lane < 16) ? 0 : 16;     // contiguous k, same as B
  const int koffB = (lane < 16) ? 0 : 16;     // contiguous k, same as A
  const int n0 = nb0 + (lane & 15);
  const int n1 = nb1 + (lane & 15);

  v8f acc0 = {};
  v8f acc1 = {};

#pragma unroll
  for (int kc = 0; kc < K; kc += 32) {
    v16bf a, b0, b1;
#pragma unroll
    for (int p = 0; p < 8; ++p) {
      const int k = kc + koffA + 2 * p;
      a[2 * p]     = A[mrow * K + k];
      a[2 * p + 1] = A[mrow * K + k + 1];
    }
#pragma unroll
    for (int j = 0; j < 16; ++j) {
      b0[j] = Wt[n0 * K + kc + koffB + j];
      b1[j] = Wt[n1 * K + kc + koffB + j];
    }
    acc0 = __builtin_amdgcn_wmma_f32_16x16x32_bf16(false, a, false, b0,
                                                   (short)0, acc0, false, false);
    acc1 = __builtin_amdgcn_wmma_f32_16x16x32_bf16(false, a, false, b1,
                                                   (short)0, acc1, false, false);
  }

  const int mshift = (lane < 16) ? 0 : 8;     // f32 C/D layout
  const float bb0 = bias[n0];
  const float bb1 = bias[n1];
#pragma unroll
  for (int r = 0; r < 8; ++r) {
    const int mm = mt * 16 + r + mshift;
    float y0 = acc0[r] + bb0;
    float y1 = acc1[r] + bb1;
    if (do_tanh) { y0 = tanhf(y0); y1 = tanhf(y1); }
    if (Hout) {
      Hout[mm * 64 + n0] = (bf16)y0;
      Hout[mm * 64 + n1] = (bf16)y1;
    } else {
      const long gr = row_base + mm;
      if (gr < rows) {
        Yout[gr * 64 + n0] = y0;
        Yout[gr * 64 + n1] = y1;
      }
    }
  }
}

// Fused 4-layer MLP: X -> tanh -> tanh -> tanh -> linear. Hidden activations
// never leave LDS.
template <int K0>
__global__ __launch_bounds__(256) void fused_mlp4_kernel(
    const bf16* __restrict__ X, long rows,
    const bf16* __restrict__ W0, const bf16* __restrict__ W1,
    const bf16* __restrict__ W2, const bf16* __restrict__ W3,
    const float* __restrict__ bias,  // [4][64]
    float* __restrict__ Y) {
  __shared__ bf16 sX[64 * K0];
  __shared__ bf16 sH[2][64 * 64];

  const long row_base = (long)blockIdx.x * 64;
  for (int i = threadIdx.x; i < 64 * K0; i += 256) {
    const int r = i / K0;
    const long gr = row_base + r;
    sX[i] = (gr < rows) ? X[gr * K0 + (i % K0)] : (bf16)0.f;
  }
  __syncthreads();
  mlp_layer<K0>(sX,    W0, bias +   0, sH[0], nullptr, 0, 0, true);
  __syncthreads();
  mlp_layer<64>(sH[0], W1, bias +  64, sH[1], nullptr, 0, 0, true);
  __syncthreads();
  mlp_layer<64>(sH[1], W2, bias + 128, sH[0], nullptr, 0, 0, true);
  __syncthreads();
  mlp_layer<64>(sH[0], W3, bias + 192, nullptr, Y, row_base, rows, false);
}

// ---------------------------------------------------------------------------
// Weight pack: f32 W[K][Nout] -> bf16 Wt[64][Kpad] (transposed, zero padded)
// ---------------------------------------------------------------------------
__global__ void pack_weights_kernel(const float* __restrict__ W,
                                    const float* __restrict__ b, int K, int Kpad,
                                    int Nout, bf16* __restrict__ Wt,
                                    float* __restrict__ bias) {
  const int i = blockIdx.x * blockDim.x + threadIdx.x;
  if (i < 64) bias[i] = (i < Nout) ? b[i] : 0.f;
  if (i >= 64 * Kpad) return;
  const int n = i / Kpad;
  const int k = i % Kpad;
  const float v = (k < K && n < Nout) ? W[k * Nout + n] : 0.f;
  Wt[i] = (bf16)v;
}

// ---------------------------------------------------------------------------
// Input assembly kernels (write bf16 with zero K-padding)
// ---------------------------------------------------------------------------
__global__ void build_edge_kernel(const float* __restrict__ Hv,
                                  const float* __restrict__ He0,
                                  const float* __restrict__ Aflat,
                                  const int* __restrict__ A0, float t,
                                  bf16* __restrict__ Xe) {
  const int i = blockIdx.x * blockDim.x + threadIdx.x;
  if (i >= RE * KEP) return;
  const int col = i % KEP;
  const int row = i / KEP;          // b*E + e
  const int e = row % Ec;
  const int b = row / Ec;
  float v;
  if (col == 0) {
    v = t;
  } else if (col < 65) {
    const int f = A0[row * 2 + 0];
    v = Hv[(b * Nc + f) * 64 + (col - 1)];
  } else if (col < 129) {
    const int to = A0[row * 2 + 1];
    v = Hv[(b * Nc + to) * 64 + (col - 65)];
  } else if (col < 193) {
    v = He0[row * 64 + (col - 129)];
  } else if (col < 199) {
    v = Aflat[(b * Ec + e) * DAc + (col - 193)];
  } else {
    v = 0.f;
  }
  Xe[i] = (bf16)v;
}

__global__ void build_bus_kernel(const float* __restrict__ Hv,
                                 const float* __restrict__ He1,
                                 const float* __restrict__ U,
                                 const float* __restrict__ Bflat, float t,
                                 bf16* __restrict__ Xb) {
  const int i = blockIdx.x * blockDim.x + threadIdx.x;
  if (i >= RB * KBP) return;
  const int col = i % KBP;
  const int row = i / KBP;          // b*N + n
  float v;
  if (col == 0)        v = t;
  else if (col < 65)   v = Hv[row * 64 + (col - 1)];
  else if (col < 129)  v = He1[row * 64 + (col - 65)];
  else if (col < 131)  v = U[row * 2 + (col - 129)];
  else if (col < 139)  v = Bflat[row * 8 + (col - 131)];
  else                 v = 0.f;
  Xb[i] = (bf16)v;
}

// ---------------------------------------------------------------------------
// State-update kernels
// ---------------------------------------------------------------------------
__global__ void hv_pre_kernel(const float* __restrict__ Hv,
                              const float* __restrict__ Vout,
                              float* __restrict__ Hnew, int n) {
  const int i = blockIdx.x * blockDim.x + threadIdx.x;
  if (i < n) Hnew[i] = Hv[i] + DTc * Vout[i];
}

__global__ void scatter_add_kernel(const float* __restrict__ P1,
                                   const float* __restrict__ P2,
                                   const int* __restrict__ A0,
                                   float* __restrict__ Hnew) {
  const int i = blockIdx.x * blockDim.x + threadIdx.x;
  if (i >= RE * 64) return;
  const int c = i & 63;
  const int be = i >> 6;            // b*E + e
  const int b = be / Ec;
  const int f  = A0[be * 2 + 0];
  const int tt = A0[be * 2 + 1];
  atomicAdd(&Hnew[(b * Nc + f) * 64 + c], P1[i]);
  atomicAdd(&Hnew[(b * Nc + tt) * 64 + c], P2[i]);
}

// Dst = Src / (||Src_row|| + 1); one wave32 per 64-wide row.
__global__ __launch_bounds__(256) void norm_rows_kernel(
    const float* __restrict__ Src, float* __restrict__ Dst, int nrows) {
  const int lane = threadIdx.x & 31;
  const int row = blockIdx.x * 8 + (threadIdx.x >> 5);
  if (row >= nrows) return;
  const float v0 = Src[row * 64 + lane];
  const float v1 = Src[row * 64 + lane + 32];
  float ss = v0 * v0 + v1 * v1;
#pragma unroll
  for (int off = 16; off; off >>= 1) ss += __shfl_xor(ss, off, 32);
  const float s = 1.f / (sqrtf(ss) + 1.f);
  Dst[row * 64 + lane]      = v0 * s;
  Dst[row * 64 + lane + 32] = v1 * s;
}

// H = norm1(H + DT*Out)
__global__ __launch_bounds__(256) void he_update_norm_kernel(
    float* __restrict__ H, const float* __restrict__ Out, int nrows) {
  const int lane = threadIdx.x & 31;
  const int row = blockIdx.x * 8 + (threadIdx.x >> 5);
  if (row >= nrows) return;
  const float v0 = H[row * 64 + lane]      + DTc * Out[row * 64 + lane];
  const float v1 = H[row * 64 + lane + 32] + DTc * Out[row * 64 + lane + 32];
  float ss = v0 * v0 + v1 * v1;
#pragma unroll
  for (int off = 16; off; off >>= 1) ss += __shfl_xor(ss, off, 32);
  const float s = 1.f / (sqrtf(ss) + 1.f);
  H[row * 64 + lane]      = v0 * s;
  H[row * 64 + lane + 32] = v1 * s;
}

__global__ void u_update_kernel(float* __restrict__ U,
                                const float* __restrict__ Out, int n) {
  const int i = blockIdx.x * blockDim.x + threadIdx.x;
  if (i < n) U[i] += DTc * Out[(i >> 1) * 64 + (i & 1)];
}

__global__ void zero_kernel(float* __restrict__ p, int n) {
  const int i = blockIdx.x * blockDim.x + threadIdx.x;
  if (i < n) p[i] = 0.f;
}

__global__ void u_init_kernel(float* __restrict__ U, int n) {
  const int i = blockIdx.x * blockDim.x + threadIdx.x;
  if (i < n) U[i] = (i & 1) ? 0.f : 1.f;
}

__global__ void copy_kernel(const float* __restrict__ s, float* __restrict__ d,
                            int n) {
  const int i = blockIdx.x * blockDim.x + threadIdx.x;
  if (i < n) d[i] = s[i];
}

// ---------------------------------------------------------------------------
// Host orchestration
// ---------------------------------------------------------------------------
static inline dim3 grid1(long n, int b) { return dim3((unsigned)((n + b - 1) / b)); }

extern "C" void kernel_launch(void* const* d_in, const int* in_sizes, int n_in,
                              void* d_out, int out_size, void* d_ws, size_t ws_size,
                              hipStream_t stream) {
  (void)in_sizes; (void)n_in; (void)out_size; (void)ws_size;

  const float* Aflat = (const float*)d_in[0];
  const float* Bflat = (const float*)d_in[1];
  const int*   A0    = (const int*)d_in[2];

  // params: phi_v00, phi_v01, phi_v10, phi_e0, phi_e1, phi_out1, each
  // [(W0,b0),(W1,b1),(W2,b2),(W3,b3)] flattened -> d_in[3 + m*8 + l*2 (+1)]
  static const int K0v[6] = {KE, KE, KB, KE, KB, KB};
  static const int K0p[6] = {KEP, KEP, KBP, KEP, KBP, KBP};

  // ---- workspace layout ----
  char* cur = (char*)d_ws;
  auto alloc = [&](size_t bytes) -> void* {
    void* r = (void*)cur;
    cur += (bytes + 255) & ~(size_t)255;
    return r;
  };

  bf16*  wt[6];
  float* wb[6];
  for (int m = 0; m < 6; ++m) {
    wt[m] = (bf16*)alloc((size_t)64 * K0p[m] * 2 + 3 * (size_t)64 * 64 * 2);
    wb[m] = (float*)alloc(4 * 64 * sizeof(float));
  }
  float* Hv     = (float*)alloc((size_t)RB * 64 * 4);
  float* He0    = (float*)alloc((size_t)RE * 64 * 4);
  float* He1    = (float*)alloc((size_t)RB * 64 * 4);
  float* U      = (float*)alloc((size_t)RB * 2 * 4);
  float* Hnew   = (float*)alloc((size_t)RB * 64 * 4);
  bf16*  Xe     = (bf16*)alloc((size_t)RE * KEP * 2);
  bf16*  Xb     = (bf16*)alloc((size_t)RB * KBP * 2);
  float* P1     = (float*)alloc((size_t)RE * 64 * 4);
  float* P2     = (float*)alloc((size_t)RE * 64 * 4);
  float* OutBus = (float*)alloc((size_t)RB * 64 * 4);

  // ---- pack weights to bf16 transposed/padded ----
  for (int m = 0; m < 6; ++m) {
    for (int l = 0; l < 4; ++l) {
      const int K    = l ? 64 : K0v[m];
      const int Kpad = l ? 64 : K0p[m];
      const int Nout = (l == 3 && m == 5) ? DUc : 64;
      const float* W = (const float*)d_in[3 + m * 8 + l * 2];
      const float* b = (const float*)d_in[3 + m * 8 + l * 2 + 1];
      bf16* dst = (l == 0) ? wt[m] : wt[m] + (size_t)64 * K0p[m] + (size_t)(l - 1) * 64 * 64;
      pack_weights_kernel<<<grid1((long)64 * Kpad, 256), 256, 0, stream>>>(
          W, b, K, Kpad, Nout, dst, wb[m] + l * 64);
    }
  }

  auto run_mlp = [&](int m, const bf16* X, long rows, float* Y) {
    const bf16* W0 = wt[m];
    const bf16* W1 = wt[m] + (size_t)64 * K0p[m];
    const bf16* W2 = W1 + 64 * 64;
    const bf16* W3 = W2 + 64 * 64;
    const dim3 g((unsigned)(rows / 64));
    if (K0p[m] == KEP)
      fused_mlp4_kernel<KEP><<<g, 256, 0, stream>>>(X, rows, W0, W1, W2, W3, wb[m], Y);
    else
      fused_mlp4_kernel<KBP><<<g, 256, 0, stream>>>(X, rows, W0, W1, W2, W3, wb[m], Y);
  };

  // ---- init state ----
  zero_kernel<<<grid1((long)RB * 64, 256), 256, 0, stream>>>(Hv, RB * 64);
  zero_kernel<<<grid1((long)RE * 64, 256), 256, 0, stream>>>(He0, RE * 64);
  zero_kernel<<<grid1((long)RB * 64, 256), 256, 0, stream>>>(He1, RB * 64);
  u_init_kernel<<<grid1((long)RB * 2, 256), 256, 0, stream>>>(U, RB * 2);

  // ---- 5 solver steps ----
  for (int s = 0; s < 5; ++s) {
    const float t = DTc * (float)s;

    build_edge_kernel<<<grid1((long)RE * KEP, 256), 256, 0, stream>>>(
        Hv, He0, Aflat, A0, t, Xe);
    run_mlp(0, Xe, RE, P1);   // phi_v00
    run_mlp(1, Xe, RE, P2);   // phi_v01

    build_bus_kernel<<<grid1((long)RB * KBP, 256), 256, 0, stream>>>(
        Hv, He1, U, Bflat, t, Xb);
    run_mlp(2, Xb, RB, OutBus);  // phi_v10

    hv_pre_kernel<<<grid1((long)RB * 64, 256), 256, 0, stream>>>(Hv, OutBus, Hnew, RB * 64);
    scatter_add_kernel<<<grid1((long)RE * 64, 256), 256, 0, stream>>>(P1, P2, A0, Hnew);
    norm_rows_kernel<<<grid1(RB, 8), 256, 0, stream>>>(Hnew, Hv, RB);

    build_bus_kernel<<<grid1((long)RB * KBP, 256), 256, 0, stream>>>(
        Hv, He1, U, Bflat, t, Xb);
    run_mlp(4, Xb, RB, OutBus);  // phi_e1
    he_update_norm_kernel<<<grid1(RB, 8), 256, 0, stream>>>(He1, OutBus, RB);

    run_mlp(3, Xe, RE, P1);      // phi_e0 (edge_in from step start)
    he_update_norm_kernel<<<grid1(RE, 8), 256, 0, stream>>>(He0, P1, RE);

    build_bus_kernel<<<grid1((long)RB * KBP, 256), 256, 0, stream>>>(
        Hv, He1, U, Bflat, t, Xb);
    run_mlp(5, Xb, RB, OutBus);  // phi_out1 (64-wide, first 2 cols valid)
    u_update_kernel<<<grid1((long)RB * 2, 256), 256, 0, stream>>>(U, OutBus, RB * 2);
  }

  copy_kernel<<<grid1((long)RB * 2, 256), 256, 0, stream>>>(U, (float*)d_out, RB * 2);
}
